// Discriminator_33036888441399
// MI455X (gfx1250) — matrix operand order repaired and checked by reading, not compiled, not run on hardware
//
#include <hip/hip_runtime.h>
#include <hip/hip_bf16.h>

typedef __bf16        v16bf __attribute__((ext_vector_type(16)));
typedef float         v8f   __attribute__((ext_vector_type(8)));
typedef unsigned int  v8u   __attribute__((ext_vector_type(8)));

#define DEV static __device__ __forceinline__

// ---- packed workspace layout ----
// bf16 weight fragments: each fragment = 512 elems (32x16 tile, lane-major,
// lane l holds 16 contiguous bf16 = its WMMA B register contents).
// Fragment order within a layer: frag = kt*NT + nt.
#define W1_OFF 0       // KT=4, NT=8 -> 16384 elems
#define W2_OFF 16384   // KT=1, NT=4 -> 2048
#define W3_OFF 18432   // 2048
#define W4_OFF 20480   // 2048
#define W5_OFF 22528   // KT=1, NT=8 -> 4096
#define WO_OFF 26624   // 4096
#define W_TOTAL 30720  // elems (61440 bytes)
#define BIAS_FLT_BASE 15360   // float index into ws (byte 61440)
// bias float offsets within bias region:
//   L1:0(128) L2:128(64) L3:192(64) L4:256(64) L5:320(128) OUT:448(128) = 576

DEV unsigned f2bfu(float f) {
  unsigned u = __float_as_uint(f);
  return (u + 0x7FFFu + ((u >> 16) & 1u)) >> 16;   // RNE fp32 -> bf16
}
DEV unsigned bf2(float a, float b) { return f2bfu(a) | (f2bfu(b) << 16); }
DEV float sigf(float x) { return 1.f / (1.f + __expf(-x)); }

DEV v8f wmma_bf16(v16bf a, v16bf b, v8f c) {
  return __builtin_amdgcn_wmma_f32_16x16x32_bf16(false, a, false, b,
                                                 (short)0, c, false, false);
}

// ------------------------------------------------------------------
// Weight/bias repack: fp32 torch layout -> WMMA B-fragment bf16 layout.
// B element e of lane l in a 32x16 tile: col n = l%16,
// K = (l<16 ? 0 : 8) + (e&7) + 16*(e>>3). Gates padded to 16-col tiles; K padded to 32.
// ------------------------------------------------------------------
__global__ void pack_kernel(
    const float* wi0, const float* wi1, const float* wi2, const float* wi3, const float* wi4,
    const float* bi0, const float* bi1, const float* bi2, const float* bi3, const float* bi4,
    const float* bh0, const float* bh1, const float* bh2, const float* bh3, const float* bh4,
    const float* wout, const float* bout,
    unsigned short* __restrict__ wsW, float* __restrict__ wsB)
{
  const float* WI[5] = {wi0, wi1, wi2, wi3, wi4};
  const float* BI[5] = {bi0, bi1, bi2, bi3, bi4};
  const float* BH[5] = {bh0, bh1, bh2, bh3, bh4};
  const int IN[5]  = {128, 32, 16, 8, 16};
  const int HS[5]  = {32, 16, 8, 16, 32};
  const int NTG[5] = {2, 1, 1, 1, 2};          // gate n-tiles (H_pad/16)
  const int WOFF[6] = {W1_OFF, W2_OFF, W3_OFF, W4_OFF, W5_OFF, WO_OFF};
  const int BOFF[6] = {0, 128, 192, 256, 320, 448};

  int t = blockIdx.x * blockDim.x + threadIdx.x;
  if (t < W_TOTAL) {
    int layer = 5;
    for (int l = 0; l < 5; ++l) { if (t < WOFF[l + 1]) { layer = l; break; } }
    int idx  = t - WOFF[layer];
    int frag = idx >> 9;
    int w    = idx & 511;
    int ln   = w >> 4;
    int e    = w & 15;
    float val = 0.f;
    if (layer < 5) {
      int NT = 4 * NTG[layer];
      int kt = frag / NT;
      int nt = frag % NT;
      int k  = kt * 32 + ((ln < 16) ? 0 : 8) + (e & 7) + ((e >> 3) * 16);
      int p  = nt * 16 + (ln & 15);
      int Hp = 16 * NTG[layer];
      int g = p / Hp, wg = p % Hp;
      if (k < IN[layer] && wg < HS[layer])
        val = WI[layer][(g * HS[layer] + wg) * IN[layer] + k];
    } else {
      int nt = frag & 7;
      int k  = ((ln < 16) ? 0 : 8) + (e & 7) + ((e >> 3) * 16);      // K = 0..31 (in = 32)
      int p  = nt * 16 + (ln & 15);           // out col 0..127
      val = wout[p * 32 + k];                 // B = w_out^T
    }
    wsW[t] = (unsigned short)f2bfu(val);
  }
  int tb = t - W_TOTAL;
  if (tb >= 0 && tb < 576) {
    float v = 0.f;
    if (tb < 448) {
      int layer = 4;
      for (int l = 0; l < 5; ++l) { if (tb < BOFF[l + 1]) { layer = l; break; } }
      int i  = tb - BOFF[layer];
      int Hp = 16 * NTG[layer];
      int g = i / Hp, wg = i % Hp;
      if (wg < HS[layer])
        v = BI[layer][g * HS[layer] + wg] + BH[layer][g * HS[layer] + wg];
    } else {
      v = bout[tb - 448];
    }
    wsB[tb] = v;
  }
}

// ------------------------------------------------------------------
// Main kernel helpers
// ------------------------------------------------------------------

// A-fragment (16x32 bf16) from a row-major 16x32 bf16 LDS tile.
// Lane l: row m = l%16; e0..7 -> K = c0..c0+7, e8..15 -> K = c0+16..c0+23,
// c0 = (l<16 ? 0 : 8).
DEV v16bf load_a_lds(const unsigned short* lh, int lane) {
  int half = lane >> 4, m = lane & 15;
  const uint4* rp = (const uint4*)(lh + m * 32);
  uint4 A0 = rp[half];
  uint4 A1 = rp[2 + half];
  v8u u;
  u[0] = A0.x; u[1] = A0.y; u[2] = A0.z; u[3] = A0.w;
  u[4] = A1.x; u[5] = A1.y; u[6] = A1.z; u[7] = A1.w;
  return __builtin_bit_cast(v16bf, u);
}

// One collapsed-LSTM layer for a 16-row tile: gates = A*B (+bias),
// h = sigmoid(o)*tanh(sigmoid(i)*tanh(g)); f-gate skipped (f*c0 == 0).
// Writes 16x32 bf16 h-tile (upper 16 cols zeroed when NTG==1) to LDS.
template <int KT, int NTG>
DEV void lstm_layer(const unsigned short* __restrict__ Wp,
                    const float* __restrict__ Bp,
                    const v16bf* afr, unsigned short* lh, int lane)
{
  const int NT = 4 * NTG, H = 16 * NTG;
  int half = lane >> 4, li = lane & 15;
  v8f acc[3 * NTG];
  const int gbase[3] = {0, 2 * NTG, 3 * NTG};   // i, g, o gate n-tile bases
#pragma unroll
  for (int gi = 0; gi < 3; ++gi) {
#pragma unroll
    for (int j = 0; j < NTG; ++j) {
      int nt = gbase[gi] + j;
      v8f c = {};
#pragma unroll
      for (int t = 0; t < KT; ++t) {
        const v16bf* bp = (const v16bf*)(Wp + (t * NT + nt) * 512);
        c = wmma_bf16(afr[t], bp[lane], c);
      }
      acc[gi * NTG + j] = c;
    }
  }
#pragma unroll
  for (int j = 0; j < NTG; ++j) {
    float bi = Bp[j * 16 + li];
    float bg = Bp[2 * H + j * 16 + li];
    float bo = Bp[3 * H + j * 16 + li];
    v8f ai = acc[0 * NTG + j], ag = acc[1 * NTG + j], ao = acc[2 * NTG + j];
#pragma unroll
    for (int s = 0; s < 8; ++s) {
      float cc = sigf(ai[s] + bi) * tanhf(ag[s] + bg);
      float hv = sigf(ao[s] + bo) * tanhf(cc);
      lh[(half * 8 + s) * 32 + j * 16 + li] = (unsigned short)f2bfu(hv);
      if (NTG == 1) lh[(half * 8 + s) * 32 + 16 + li] = 0;  // zero-pad K 16..31
    }
  }
}

__global__ void __launch_bounds__(256, 2) lstm_stack_kernel(
    const float* __restrict__ x, const unsigned short* __restrict__ wsW,
    const float* __restrict__ wsB, float* __restrict__ out, int nTiles)
{
  __shared__ __align__(16) unsigned short lds_h[8][16 * 32];  // 1KB / wave
  int lane = threadIdx.x & 31;
  int wIn  = threadIdx.x >> 5;
  unsigned short* lh = lds_h[wIn];
  int wave = blockIdx.x * (blockDim.x >> 5) + wIn;
  int nW   = gridDim.x * (blockDim.x >> 5);
  int half = lane >> 4, li = lane & 15;

  for (int tile = wave; tile < nTiles; tile += nW) {
    size_t r0 = (size_t)tile * 16;

    // prefetch next tile of x
    int ntile = tile + nW;
    if (ntile < nTiles)
      __builtin_prefetch(x + ((size_t)ntile * 16 + li) * 128 + half * 8, 0, 1);

    // ---- layer 1: A fragments straight from x (fp32 -> bf16) ----
    v16bf a1[4];
    const float* xr = x + (r0 + li) * 128 + half * 8;
#pragma unroll
    for (int t = 0; t < 4; ++t) {
      const float* p0 = xr + t * 32;
      v8u u;
#pragma unroll
      for (int j = 0; j < 4; ++j) u[j] = bf2(p0[2 * j], p0[2 * j + 1]);
#pragma unroll
      for (int j = 0; j < 4; ++j) u[4 + j] = bf2(p0[16 + 2 * j], p0[16 + 2 * j + 1]);
      a1[t] = __builtin_bit_cast(v16bf, u);
    }

    lstm_layer<4, 2>(wsW + W1_OFF, wsB + 0, a1, lh, lane);
    asm volatile("s_wait_dscnt 0x0" ::: "memory");
    v16bf a2 = load_a_lds(lh, lane);

    lstm_layer<1, 1>(wsW + W2_OFF, wsB + 128, &a2, lh, lane);
    asm volatile("s_wait_dscnt 0x0" ::: "memory");
    v16bf a3 = load_a_lds(lh, lane);

    lstm_layer<1, 1>(wsW + W3_OFF, wsB + 192, &a3, lh, lane);
    asm volatile("s_wait_dscnt 0x0" ::: "memory");
    v16bf a4 = load_a_lds(lh, lane);

    lstm_layer<1, 1>(wsW + W4_OFF, wsB + 256, &a4, lh, lane);
    asm volatile("s_wait_dscnt 0x0" ::: "memory");
    v16bf a5 = load_a_lds(lh, lane);

    lstm_layer<1, 2>(wsW + W5_OFF, wsB + 320, &a5, lh, lane);
    asm volatile("s_wait_dscnt 0x0" ::: "memory");
    v16bf a6 = load_a_lds(lh, lane);

    // ---- output layer: [16x32] x [32x128] + b_out, fp32 out ----
    const float* bo = wsB + 448;
    float* orow = out + (r0 + half * 8) * 128;
#pragma unroll
    for (int nt = 0; nt < 8; ++nt) {
      const v16bf* bp = (const v16bf*)(wsW + WO_OFF + nt * 512);
      v8f c = {};
      c = wmma_bf16(a6, bp[lane], c);
      float bb = bo[nt * 16 + li];
#pragma unroll
      for (int s = 0; s < 8; ++s)
        orow[(size_t)s * 128 + nt * 16 + li] = c[s] + bb;
    }
  }
}

// ------------------------------------------------------------------
// Launch
// ------------------------------------------------------------------
extern "C" void kernel_launch(void* const* d_in, const int* in_sizes, int n_in,
                              void* d_out, int out_size, void* d_ws, size_t ws_size,
                              hipStream_t stream) {
  // inputs: x, w_ih[0..4], w_hh[0..4] (unused: h0==0), b_ih[0..4], b_hh[0..4],
  //         w_out, b_out
  const float* x = (const float*)d_in[0];
  const float* wi[5], *bi[5], *bh[5];
  for (int l = 0; l < 5; ++l) {
    wi[l] = (const float*)d_in[1 + l];
    bi[l] = (const float*)d_in[11 + l];
    bh[l] = (const float*)d_in[16 + l];
  }
  const float* wout = (const float*)d_in[21];
  const float* bout = (const float*)d_in[22];

  unsigned short* wsW = (unsigned short*)d_ws;
  float* wsB = (float*)d_ws + BIAS_FLT_BASE;

  const int packTotal = W_TOTAL + 576;
  pack_kernel<<<(packTotal + 255) / 256, 256, 0, stream>>>(
      wi[0], wi[1], wi[2], wi[3], wi[4],
      bi[0], bi[1], bi[2], bi[3], bi[4],
      bh[0], bh[1], bh[2], bh[3], bh[4],
      wout, bout, wsW, wsB);

  const int nRows  = 256 * 2048;
  const int nTiles = nRows / 16;          // 32768 tiles of 16 rows
  lstm_stack_kernel<<<512, 256, 0, stream>>>(x, wsW, wsB, (float*)d_out, nTiles);
}
